// BEVMulti_Fuser_79937931313272
// MI455X (gfx1250) — matrix operand order repaired and checked by reading, not compiled, not run on hardware
//
#include <hip/hip_runtime.h>

#define LV 5
#define NHD 4
#define PP 4
#define EE 256
#define CC 256
#define HB 128
#define WB 128
#define NQ (HB*WB)
#define HD (EE/NHD)
#define NL 6
#define FF (2*EE)

typedef _Float16 half_t;
typedef __attribute__((ext_vector_type(2)))  _Float16 v2h;
typedef __attribute__((ext_vector_type(8)))  _Float16 v8h;
typedef __attribute__((ext_vector_type(16))) _Float16 v16h;
typedef __attribute__((ext_vector_type(8)))  float    v8f;
typedef __attribute__((ext_vector_type(4)))  unsigned int u32x4;
typedef __attribute__((ext_vector_type(8)))  int          i32x8;
typedef __attribute__((ext_vector_type(4)))  int          i32x4;

// A fragment: 16 contiguous K values per lane, halves[0..15]=K kb..kb+15
__device__ __forceinline__ v16h frag_a(const half_t* p) {
    v8h lo = *(const v8h*)p;
    v8h hi = *(const v8h*)(p + 8);
    return __builtin_shufflevector(lo, hi, 0,1,2,3,4,5,6,7,8,9,10,11,12,13,14,15);
}
// B fragment: 16 contiguous K values per lane (two 16B loads; LDS base only 16B-aligned)
__device__ __forceinline__ v16h frag_b(const half_t* p) {
    v8h lo = *(const v8h*)p;
    v8h hi = *(const v8h*)(p + 8);
    return __builtin_shufflevector(lo, hi, 0,1,2,3,4,5,6,7,8,9,10,11,12,13,14,15);
}

// ---------------- WMMA GEMM: C[M,N] = A[M,K] @ Bt[N,K]^T + bias ------------------------
// Wave computes a 16x(NS*16) tile, NS static. flags: 1=f16 store, 2=relu, 4=+res, 8=TDM-stage B.
template <int NS>
__global__ void k_gemm_wmma(const half_t* __restrict__ A, int lda,
                            const half_t* __restrict__ Bt, int ldb,
                            float* __restrict__ Cf, half_t* __restrict__ Ch, int ldc,
                            int M, int N, int K,
                            const float* __restrict__ bias,
                            const float* __restrict__ res,
                            int flags, int n0)
{
    extern __shared__ char smem_raw[];
    half_t* sB = (half_t*)smem_raw;

    const int t       = threadIdx.x & 31;
    const int wave    = threadIdx.x >> 5;
    const int tile_n64= n0 + blockIdx.x * 64;
    const int tile_m  = (blockIdx.y * (blockDim.x >> 5) + wave) * 16;
    const int r16     = t & 15;
    const int kb8     = (t >> 4) * 8;
    const int kb16    = (t >> 4) * 16;
    const int rows    = NS * 16;            // B rows staged (always fully in-bounds)

    if (flags & 8) {
        if (wave == 0) {
#if defined(__has_builtin) && __has_builtin(__builtin_amdgcn_tensor_load_to_lds)
            // Tensor DMA: 2-D tile [rows x K] of 2-byte elements, Bt block -> LDS (ISA 8.3-8.6)
            unsigned long long ga = (unsigned long long)(Bt + (size_t)tile_n64 * ldb);
            unsigned int lds_off  = (unsigned int)(unsigned long long)(void*)sB;
            u32x4 g0;
            g0[0] = 1u;                                        // count=1, user descriptor
            g0[1] = lds_off;                                   // lds_addr (bytes)
            g0[2] = (unsigned int)(ga & 0xffffffffull);        // global_addr lo
            g0[3] = (unsigned int)((ga >> 32) & 0x1ffffffull)  // global_addr hi (bits 120:96)
                  | (2u << 30);                                // type=2 ("image")
            i32x8 g1;
            g1[0] = 0x00010000;                                // workgroup_mask=0, data_size=1 (2B)
            g1[1] = (int)(((unsigned)K & 0xffffu) << 16);      // tensor_dim0[15:0]
            g1[2] = (int)((((unsigned)K >> 16) & 0xffffu)      // tensor_dim0[31:16]
                  | (((unsigned)rows & 0xffffu) << 16));       // tensor_dim1[15:0]
            g1[3] = (int)(((unsigned)K & 0xffffu) << 16);      // tensor_dim1[31:16]=0, tile_dim0=K
            g1[4] = rows;                                      // tile_dim1=rows, tile_dim2=0
            g1[5] = ldb;                                       // tensor_dim0_stride[31:0]
            g1[6] = 0;                                         // stride hi, tensor_dim1_stride lo
            g1[7] = 0;
            i32x4 z4 = {0, 0, 0, 0};
            i32x8 z8 = {0, 0, 0, 0, 0, 0, 0, 0};
            __builtin_amdgcn_tensor_load_to_lds(g0, g1, z4, z4, z8, 0);
            __builtin_amdgcn_s_wait_tensorcnt(0);
#else
            // cooperative fallback copy (wave 0 only)
            int totals = rows * (K >> 3);
            for (int idx = t; idx < totals; idx += 32) {
                int rr = idx / (K >> 3);
                int kk = (idx - rr * (K >> 3)) << 3;
                *(v8h*)(sB + (size_t)rr * K + kk) =
                    *(const v8h*)(Bt + (size_t)(tile_n64 + rr) * ldb + kk);
            }
#endif
        }
        __syncthreads();
    }

    const half_t* arow = A + (size_t)(tile_m + r16) * lda + kb16;
    v8f c[NS];
#pragma unroll
    for (int s = 0; s < NS; ++s) c[s] = (v8f){};

    if (flags & 8) {
        const half_t* bbase = sB + (size_t)r16 * K + kb16;
#pragma unroll 2
        for (int k = 0; k < K; k += 32) {
            __builtin_prefetch((const void*)(arow + k + 256), 0, 0);
            v16h a = frag_a(arow + k);
#pragma unroll
            for (int s = 0; s < NS; ++s) {
                v16h b = frag_b(bbase + (size_t)s * 16 * K + k);
                c[s] = __builtin_amdgcn_wmma_f32_16x16x32_f16(false, a, false, b,
                                                              (short)0, c[s], false, false);
            }
        }
    } else {
        const half_t* bbase = Bt + ((size_t)tile_n64 + r16) * ldb + kb16;
#pragma unroll 2
        for (int k = 0; k < K; k += 32) {
            __builtin_prefetch((const void*)(arow + k + 256), 0, 0);
            v16h a = frag_a(arow + k);
#pragma unroll
            for (int s = 0; s < NS; ++s) {
                v16h b = frag_b(bbase + (size_t)s * 16 * ldb + k);
                c[s] = __builtin_amdgcn_wmma_f32_16x16x32_f16(false, a, false, b,
                                                              (short)0, c[s], false, false);
            }
        }
    }

    const int m0 = tile_m + ((t >> 4) * 8);
#pragma unroll
    for (int s = 0; s < NS; ++s) {
        const int n_g = tile_n64 + s * 16 + r16;
        const float bv = bias ? bias[n_g] : 0.0f;
#pragma unroll
        for (int r = 0; r < 8; ++r) {
            const int m_g = m0 + r;
            float v = c[s][r] + bv;
            if (flags & 4) v += res[(size_t)m_g * ldc + n_g];
            if (flags & 2) v = v > 0.0f ? v : 0.0f;
            if (flags & 1) Ch[(size_t)m_g * ldc + n_g] = (half_t)v;
            else           Cf[(size_t)m_g * ldc + n_g] = v;
        }
    }
}

// ---------------- weight transpose + f32->f16: dst[n*K+k] = src[k*N+n] ----------------
__global__ void k_w2h_t(const float* __restrict__ src, half_t* __restrict__ dst, int K, int N)
{
    int i = blockIdx.x * blockDim.x + threadIdx.x;
    if (i >= K * N) return;
    int k = i / N, n = i - k * N;
    dst[(size_t)n * K + k] = (half_t)src[i];
}

// ---------------- per-level channel LN of feat_bev -> fh[n*1280 + l*256 + c] (f16) -----
__global__ void k_norm0(const float* __restrict__ feat, const float* __restrict__ g,
                        const float* __restrict__ b, half_t* __restrict__ fh)
{
    int row = blockIdx.x * (blockDim.x >> 5) + (threadIdx.x >> 5); // l*NQ + n
    int t = threadIdx.x & 31;
    int l = row >> 14;
    int n = row & (NQ - 1);
    const float* src = feat + (size_t)l * CC * NQ + n;
    float x[8]; float s = 0.f;
#pragma unroll
    for (int j = 0; j < 8; ++j) { x[j] = src[(size_t)(t + 32*j) * NQ]; s += x[j]; }
#pragma unroll
    for (int o = 16; o; o >>= 1) s += __shfl_xor(s, o, 32);
    float mean = s * (1.0f / CC);
    float q2 = 0.f;
#pragma unroll
    for (int j = 0; j < 8; ++j) { float d = x[j] - mean; q2 += d * d; }
#pragma unroll
    for (int o = 16; o; o >>= 1) q2 += __shfl_xor(q2, o, 32);
    float rinv = rsqrtf(q2 * (1.0f / CC) + 1e-5f);
    half_t* dst = fh + (size_t)n * (LV * CC) + l * CC;
#pragma unroll
    for (int j = 0; j < 8; ++j) {
        int cidx = t + 32 * j;
        dst[cidx] = (half_t)((x[j] - mean) * rinv * g[l*CC + cidx] + b[l*CC + cidx]);
    }
}

// ---------------- qp = (half)(q + pos) -------------------------------------------------
__global__ void k_qp(const float* __restrict__ q, const float* __restrict__ pos_row,
                     const float* __restrict__ pos_col, half_t* __restrict__ qp)
{
    int i = blockIdx.x * blockDim.x + threadIdx.x;   // n*256 + e
    int e = i & 255, n = i >> 8;
    int w = n & (WB - 1), h = n >> 7;
    float pv = (e < 128) ? pos_col[w * 128 + e] : pos_row[h * 128 + (e - 128)];
    qp[i] = (half_t)(q[i] + pv);
}

// ---------------- softmax over 20 per (n,h) --------------------------------------------
__global__ void k_softmax(float* __restrict__ aw)
{
    int idx = blockIdx.x * blockDim.x + threadIdx.x; // n*4 + h
    if (idx >= NQ * NHD) return;
    float* p = aw + (size_t)(idx >> 2) * (NHD * LV * PP) + (idx & 3) * (LV * PP);
    float mx = -1e30f;
    float e[LV * PP];
#pragma unroll
    for (int j = 0; j < LV * PP; ++j) mx = fmaxf(mx, p[j]);
    float s = 0.f;
#pragma unroll
    for (int j = 0; j < LV * PP; ++j) { e[j] = __expf(p[j] - mx); s += e[j]; }
    float inv = 1.0f / s;
#pragma unroll
    for (int j = 0; j < LV * PP; ++j) p[j] = e[j] * inv;
}

// ---------------- deformable bilinear sampling: wave per (n,h), lane owns 2 dims -------
__global__ void k_sample(const float* __restrict__ off, const float* __restrict__ aw,
                         const half_t* __restrict__ val, half_t* __restrict__ acch)
{
    int wid = blockIdx.x * (blockDim.x >> 5) + (threadIdx.x >> 5); // n*4 + h
    int t = threadIdx.x & 31;
    int n = wid >> 2, h = wid & 3;
    int wi = n & (WB - 1), hi = n >> 7;
    float bx = wi + 0.5f;   // refx * W
    float by = hi + 0.5f;   // refy * H
    const float* op = off + (size_t)n * (NHD*LV*PP*2) + h * (LV*PP*2);
    const float* ap = aw  + (size_t)n * (NHD*LV*PP)   + h * (LV*PP);
    float a0 = 0.f, a1 = 0.f;  // dims d = 2t, 2t+1 of this head
#pragma unroll
    for (int l = 0; l < LV; ++l) {
        const half_t* vb = val + (size_t)l * NQ * EE + h * HD + 2 * t;
#pragma unroll
        for (int p = 0; p < PP; ++p) {
            int j = l * PP + p;
            float x = bx + op[j*2 + 0] - 0.5f;
            float y = by + op[j*2 + 1] - 0.5f;
            float x0f = floorf(x), y0f = floorf(y);
            float fx = x - x0f, fy = y - y0f;
            int x0 = (int)x0f, y0 = (int)y0f;
            float wgt = ap[j];
            float w00 = (1.f-fx)*(1.f-fy)*wgt, w10 = fx*(1.f-fy)*wgt;
            float w01 = (1.f-fx)*fy*wgt,       w11 = fx*fy*wgt;
            auto corner = [&](int ix, int iy, float wv) {
                if (ix >= 0 && ix < WB && iy >= 0 && iy < HB) {
                    v2h d = *(const v2h*)(vb + (size_t)(iy * WB + ix) * EE);
                    a0 += wv * (float)d[0];
                    a1 += wv * (float)d[1];
                }
            };
            corner(x0,   y0,   w00);
            corner(x0+1, y0,   w10);
            corner(x0,   y0+1, w01);
            corner(x0+1, y0+1, w11);
        }
    }
    v2h o; o[0] = (half_t)a0; o[1] = (half_t)a1;
    *(v2h*)(acch + (size_t)n * EE + h * HD + 2 * t) = o;
}

// ---------------- LayerNorm over E=256, wave per row; optional f16 copy ----------------
__global__ void k_ln(float* __restrict__ x, const float* __restrict__ g,
                     const float* __restrict__ b, half_t* __restrict__ xh)
{
    int row = blockIdx.x * (blockDim.x >> 5) + (threadIdx.x >> 5);
    int t = threadIdx.x & 31;
    float* p = x + (size_t)row * EE;
    float v[8]; float s = 0.f;
#pragma unroll
    for (int j = 0; j < 8; ++j) { v[j] = p[t*8 + j]; s += v[j]; }
#pragma unroll
    for (int o = 16; o; o >>= 1) s += __shfl_xor(s, o, 32);
    float mean = s * (1.0f / EE);
    float q2 = 0.f;
#pragma unroll
    for (int j = 0; j < 8; ++j) { float d = v[j] - mean; q2 += d * d; }
#pragma unroll
    for (int o = 16; o; o >>= 1) q2 += __shfl_xor(q2, o, 32);
    float rinv = rsqrtf(q2 * (1.0f / EE) + 1e-5f);
#pragma unroll
    for (int j = 0; j < 8; ++j) {
        int e = t * 8 + j;
        float y = (v[j] - mean) * rinv * g[e] + b[e];
        p[e] = y;
        if (xh) xh[(size_t)row * EE + e] = (half_t)y;
    }
}

// ---------------- final transpose: out[e*NQ + n] = q[n*E + e] --------------------------
__global__ void k_out_t(const float* __restrict__ q, float* __restrict__ out)
{
    int i = blockIdx.x * blockDim.x + threadIdx.x; // n*256 + e
    int e = i & 255, n = i >> 8;
    out[(size_t)e * NQ + n] = q[i];
}

// =======================================================================================
extern "C" void kernel_launch(void* const* d_in, const int* in_sizes, int n_in,
                              void* d_out, int out_size, void* d_ws, size_t ws_size,
                              hipStream_t stream)
{
    const float* feat_bev = (const float*)d_in[0];
    const float* norm0_g  = (const float*)d_in[1];
    const float* norm0_b  = (const float*)d_in[2];
    const float* in_w     = (const float*)d_in[3];
    const float* in_b     = (const float*)d_in[4];
    const float* pos_row  = (const float*)d_in[5];
    const float* pos_col  = (const float*)d_in[6];
    const float* off_w    = (const float*)d_in[7];
    const float* off_b    = (const float*)d_in[8];
    const float* aw_w     = (const float*)d_in[9];
    const float* aw_b     = (const float*)d_in[10];
    const float* val_w    = (const float*)d_in[11];
    const float* val_b    = (const float*)d_in[12];
    const float* out_w    = (const float*)d_in[13];
    const float* out_b    = (const float*)d_in[14];
    const float* ln1_g    = (const float*)d_in[15];
    const float* ln1_b    = (const float*)d_in[16];
    const float* ln2_g    = (const float*)d_in[17];
    const float* ln2_b    = (const float*)d_in[18];
    const float* ffn_w1   = (const float*)d_in[19];
    const float* ffn_b1   = (const float*)d_in[20];
    const float* ffn_w2   = (const float*)d_in[21];
    const float* ffn_b2   = (const float*)d_in[22];

    // ---- workspace carve-up (256B aligned) ----
    char* ws = (char*)d_ws;
    size_t cur = 0;
    auto take = [&](size_t bytes) -> char* {
        char* p = ws + cur;
        cur += (bytes + 255) & ~size_t(255);
        return p;
    };
    half_t* fh    = (half_t*)take((size_t)NQ * LV * CC * 2);   // normed feats [NQ, L*C]
    half_t* in_wT = (half_t*)take((size_t)EE * LV * CC * 2);
    half_t* offT  = (half_t*)take((size_t)NL * 160 * EE * 2);
    half_t* awT   = (half_t*)take((size_t)NL * 80 * EE * 2);
    half_t* valT  = (half_t*)take((size_t)NL * EE * EE * 2);
    half_t* outT  = (half_t*)take((size_t)NL * EE * EE * 2);
    half_t* f1T   = (half_t*)take((size_t)NL * FF * EE * 2);
    half_t* f2T   = (half_t*)take((size_t)NL * EE * FF * 2);
    float*  qbuf  = (float*) take((size_t)NQ * EE * 4);
    half_t* qph   = (half_t*)take((size_t)NQ * EE * 2);
    half_t* qh    = (half_t*)take((size_t)NQ * EE * 2);
    float*  offb  = (float*) take((size_t)NQ * 160 * 4);
    float*  awb   = (float*) take((size_t)NQ * 80 * 4);
    half_t* valb  = (half_t*)take((size_t)LV * NQ * EE * 2);
    half_t* acch  = (half_t*)take((size_t)NQ * EE * 2);
    half_t* h1    = (half_t*)take((size_t)NQ * FF * 2);
    (void)ws_size; (void)in_sizes; (void)n_in; (void)out_size;

    // ---- launch helpers ----
    auto tconv = [&](const float* src, half_t* dst, int K, int N) {
        int n = K * N;
        k_w2h_t<<<dim3((n + 255) / 256), dim3(256), 0, stream>>>(src, dst, K, N);
    };
    auto gemm = [&](const half_t* A, int lda, const half_t* Bt, int ldb,
                    float* Cf, half_t* Ch, int ldc, int M, int N, int K,
                    const float* bias, const float* res, int flags) {
        int f = flags;
        bool lds = (K <= 512);
        if (lds) f |= 8;
        const int gy   = M / 128;
        const int full = N / 64;
        const int rem  = (N - full * 64) / 16;     // 0..3 remaining 16-wide subtiles
        if (full > 0) {
            size_t sh = lds ? (size_t)64 * K * sizeof(half_t) : 0;
            k_gemm_wmma<4><<<dim3(full, gy), dim3(256), sh, stream>>>(
                A, lda, Bt, ldb, Cf, Ch, ldc, M, N, K, bias, res, f, 0);
        }
        if (rem) {
            size_t sh = lds ? (size_t)rem * 16 * K * sizeof(half_t) : 0;
            dim3 g1(1, gy);
            if (rem == 1)
                k_gemm_wmma<1><<<g1, dim3(256), sh, stream>>>(
                    A, lda, Bt, ldb, Cf, Ch, ldc, M, N, K, bias, res, f, full * 64);
            else if (rem == 2)
                k_gemm_wmma<2><<<g1, dim3(256), sh, stream>>>(
                    A, lda, Bt, ldb, Cf, Ch, ldc, M, N, K, bias, res, f, full * 64);
            else
                k_gemm_wmma<3><<<g1, dim3(256), sh, stream>>>(
                    A, lda, Bt, ldb, Cf, Ch, ldc, M, N, K, bias, res, f, full * 64);
        }
    };

    // ---- weight prep (f32 -> f16 transposed [N,K]) ----
    tconv(in_w, in_wT, LV * CC, EE);
    for (int i = 0; i < NL; ++i) {
        tconv(off_w  + (size_t)i * EE * 160, offT + (size_t)i * 160 * EE, EE, 160);
        tconv(aw_w   + (size_t)i * EE * 80,  awT  + (size_t)i * 80 * EE,  EE, 80);
        tconv(val_w  + (size_t)i * CC * EE,  valT + (size_t)i * EE * EE,  CC, EE);
        tconv(out_w  + (size_t)i * EE * EE,  outT + (size_t)i * EE * EE,  EE, EE);
        tconv(ffn_w1 + (size_t)i * EE * FF,  f1T  + (size_t)i * FF * EE,  EE, FF);
        tconv(ffn_w2 + (size_t)i * FF * EE,  f2T  + (size_t)i * EE * FF,  FF, EE);
    }

    // ---- per-level LN of feat_bev -> fh [NQ, L*C] f16 ----
    k_norm0<<<dim3(LV * NQ / 8), dim3(256), 0, stream>>>(feat_bev, norm0_g, norm0_b, fh);

    // ---- input projection: q = fh @ in_w + in_b  (M=NQ, K=1280, N=256; direct-global B) ----
    gemm(fh, LV * CC, in_wT, LV * CC, qbuf, nullptr, EE, NQ, EE, LV * CC, in_b, nullptr, 0);

    // ---- layers ----
    for (int i = 0; i < NL; ++i) {
        k_qp<<<dim3(NQ * EE / 256), dim3(256), 0, stream>>>(qbuf, pos_row, pos_col, qph);
        gemm(qph, EE, offT + (size_t)i * 160 * EE, EE, offb, nullptr, 160,
             NQ, 160, EE, off_b + (size_t)i * 160, nullptr, 0);
        gemm(qph, EE, awT + (size_t)i * 80 * EE, EE, awb, nullptr, 80,
             NQ, 80, EE, aw_b + (size_t)i * 80, nullptr, 0);
        k_softmax<<<dim3(NQ * NHD / 256), dim3(256), 0, stream>>>(awb);
        for (int l = 0; l < LV; ++l) {
            gemm(fh + (size_t)l * CC, LV * CC, valT + (size_t)i * EE * EE, EE,
                 nullptr, valb + (size_t)l * NQ * EE, EE,
                 NQ, EE, CC, val_b + (size_t)i * EE, nullptr, 1);
        }
        k_sample<<<dim3(NQ * NHD / 8), dim3(256), 0, stream>>>(offb, awb, valb, acch);
        gemm(acch, EE, outT + (size_t)i * EE * EE, EE, qbuf, nullptr, EE,
             NQ, EE, EE, out_b + (size_t)i * EE, qbuf, 4);
        k_ln<<<dim3(NQ / 8), dim3(256), 0, stream>>>(qbuf, ln1_g + (size_t)i * EE,
                                                     ln1_b + (size_t)i * EE, qh);
        gemm(qh, EE, f1T + (size_t)i * FF * EE, EE, nullptr, h1, FF,
             NQ, FF, EE, ffn_b1 + (size_t)i * FF, nullptr, 1 | 2);
        gemm(h1, FF, f2T + (size_t)i * EE * FF, FF, qbuf, nullptr, EE,
             NQ, EE, FF, ffn_b2 + (size_t)i * EE, qbuf, 4);
        k_ln<<<dim3(NQ / 8), dim3(256), 0, stream>>>(qbuf, ln2_g + (size_t)i * EE,
                                                     ln2_b + (size_t)i * EE, nullptr);
    }

    // ---- output transpose [NQ,E] -> [E,H,W] ----
    k_out_t<<<dim3(NQ * EE / 256), dim3(256), 0, stream>>>(qbuf, (float*)d_out);
}
